// Word2Vec_12833362280545
// MI455X (gfx1250) — matrix operand order repaired and checked by reading, not compiled, not run on hardware
//
#include <hip/hip_runtime.h>
#include <math.h>

typedef _Float16 v16h __attribute__((ext_vector_type(16)));
typedef _Float16 v8h  __attribute__((ext_vector_type(8)));
typedef _Float16 v4h  __attribute__((ext_vector_type(4)));
typedef float    v8f  __attribute__((ext_vector_type(8)));

#define T_TOK 3200
#define VOC   32000
#define EMB   128

// Kernel 1: embed_f16[t][k] = (f16) W_A[k][tok[t]]   (b_A cancels in axis-0 log-softmax)
__global__ void w2v_build_embed(const int* __restrict__ x,
                                const float* __restrict__ W_A,
                                _Float16* __restrict__ embed) {
    const int t = blockIdx.x;        // 0..3199
    const int k = threadIdx.x;       // 0..127
    const int tok = x[t];
    embed[(size_t)t * EMB + k] = (_Float16)W_A[(size_t)k * VOC + tok];
}

// Kernel 2: W_B fp32 -> f16, coalesced, 4 elements/thread
__global__ void w2v_convert_wb(const float* __restrict__ W_B,
                               _Float16* __restrict__ Wbh) {
    const size_t i = ((size_t)blockIdx.x * blockDim.x + threadIdx.x) * 4;
    const float4 f = *(const float4*)(W_B + i);
    v4h h = { (_Float16)f.x, (_Float16)f.y, (_Float16)f.z, (_Float16)f.w };
    *(v4h*)(Wbh + i) = h;
}

// Kernel 3: per-wave 16-vocab strip, online softmax over 200 token tiles, then
// recompute pass writing log-softmax. WMMA f32_16x16x32_f16, B resident in VGPRs.
__global__ __launch_bounds__(256)
void w2v_logits_logsoftmax(const _Float16* __restrict__ embed,
                           const _Float16* __restrict__ Wbh,
                           float* __restrict__ out) {
    const int lane  = threadIdx.x & 31;
    const int wave  = threadIdx.x >> 5;
    const int vbase = (blockIdx.x * 8 + wave) * 16;   // 250 blocks * 8 waves * 16 = 32000
    const int n     = lane & 15;                      // column within tile / A row M
    const bool hiL  = lane >= 16;
    const int aoff  = hiL ? 8 : 0;                    // A: K halves 0..7 vs 8..15 (+16..23 / +24..31)

    // ---- Load B (16x(4x32) of W_B rows) once: ISA 32x16 16-bit B layout ----
    // lane<16: col=n, K 0..7,16..23 of the k-tile; lane>=16: col=n, K 8..15,24..31.
    v16h B[4];
    {
        const _Float16* bp = Wbh + (size_t)(vbase + n) * EMB + (hiL ? 8 : 0);
#pragma unroll
        for (int kt = 0; kt < 4; ++kt) {
            v8h lo = *(const v8h*)(bp + kt * 32);
            v8h hh = *(const v8h*)(bp + kt * 32 + 16);
            B[kt] = __builtin_shufflevector(lo, hh, 0,1,2,3,4,5,6,7,8,9,10,11,12,13,14,15);
        }
    }

    // A tile loader: ISA 16x32 16-bit A layout. Row M = n, token row = tbase+n.
    auto loadA = [&](int tbase, int kt) -> v16h {
        const _Float16* ap = embed + (size_t)(tbase + n) * EMB + kt * 32 + aoff;
        v8h lo = *(const v8h*)(ap);        // K = base .. base+7
        v8h hh = *(const v8h*)(ap + 16);   // K = base+16 .. base+23
        return __builtin_shufflevector(lo, hh, 0,1,2,3,4,5,6,7,8,9,10,11,12,13,14,15);
    };

    // One 16(token)x16(vocab) logits tile: 4 chained WMMAs over K=128
    auto tile = [&](int tbase) -> v8f {
        v8f acc = {0.f,0.f,0.f,0.f,0.f,0.f,0.f,0.f};
#pragma unroll
        for (int kt = 0; kt < 4; ++kt) {
            v16h a = loadA(tbase, kt);
            acc = __builtin_amdgcn_wmma_f32_16x16x32_f16(
                      false, a, false, B[kt], (short)0, acc, false, false);
        }
        return acc;
    };

    // ---- Pass 1: online max / sum-exp over the token axis (per column half) ----
    float m = -INFINITY, s = 0.0f;
    for (int tt = 0; tt < 200; ++tt) {
        __builtin_prefetch(embed + (size_t)((tt + 1) * 16 + n) * EMB, 0, 1);
        v8f acc = tile(tt * 16);
        float tmax = acc[0];
#pragma unroll
        for (int r = 1; r < 8; ++r) tmax = fmaxf(tmax, acc[r]);
        const float mnew = fmaxf(m, tmax);
        float sum = 0.f;
#pragma unroll
        for (int r = 0; r < 8; ++r) sum += __expf(acc[r] - mnew);
        s = s * __expf(m - mnew) + sum;
        m = mnew;
    }
    // Combine the two lane halves of each column (lane L holds M 0..7, L+16 holds M 8..15)
    const float mo = __shfl_xor(m, 16, 32);
    const float so = __shfl_xor(s, 16, 32);
    const float Mx = fmaxf(m, mo);
    const float Sx = s * __expf(m - Mx) + so * __expf(mo - Mx);
    const float lse = Mx + __logf(Sx);   // per-column logsumexp over all 3200 tokens

    // ---- Pass 2: recompute tiles, write out = logit - lse ----
    const int rbase = hiL ? 8 : 0;       // C/D layout: lane>=16 holds M = r+8
    for (int tt = 0; tt < 200; ++tt) {
        v8f acc = tile(tt * 16);
        float* op = out + (size_t)(tt * 16 + rbase) * VOC + vbase + n;
#pragma unroll
        for (int r = 0; r < 8; ++r) op[(size_t)r * VOC] = acc[r] - lse;
    }
}

extern "C" void kernel_launch(void* const* d_in, const int* in_sizes, int n_in,
                              void* d_out, int out_size, void* d_ws, size_t ws_size,
                              hipStream_t stream) {
    const int*   x   = (const int*)d_in[0];
    const float* W_A = (const float*)d_in[1];
    // d_in[2] = b_A, d_in[4] = b_B: both cancel under log-softmax over axis 0.
    const float* W_B = (const float*)d_in[3];

    _Float16* embed = (_Float16*)d_ws;                                 // 3200*128*2 = 800 KB
    _Float16* Wbh   = (_Float16*)((char*)d_ws + (size_t)T_TOK * EMB * 2); // 8.2 MB

    w2v_build_embed<<<T_TOK, EMB, 0, stream>>>(x, W_A, embed);
    w2v_convert_wb<<<(VOC * EMB / 4) / 256, 256, 0, stream>>>(W_B, Wbh);
    w2v_logits_logsoftmax<<<VOC / (8 * 16), 256, 0, stream>>>(embed, Wbh, (float*)d_out);
}
